// LearnedCitationSelector_55078660604304
// MI455X (gfx1250) — matrix operand order repaired and checked by reading, not compiled, not run on hardware
//
#include <hip/hip_runtime.h>
#include <hip/hip_bf16.h>
#include <math.h>

#define TEXT_DIM 768
#define HIDDEN   256
#define BB       16
#define NN       4096
#define TOPK     64

// ---- CDNA5 WMMA types ----
typedef __attribute__((ext_vector_type(16))) __bf16    v16bf;
typedef __attribute__((ext_vector_type(8)))  float     v8f;
typedef __attribute__((ext_vector_type(4)))  unsigned  u32x4;

union BF16x16 { v16bf v; u32x4 q[2]; };

__device__ __forceinline__ unsigned short f2bf(float f) {
  unsigned int u = __builtin_bit_cast(unsigned int, f);
  u = (u + 0x7FFFu + ((u >> 16) & 1u)) >> 16;   // round-to-nearest-even
  return (unsigned short)u;
}

// ---- workspace layout (float indices), then bf16 M matrix ----
enum {
  WS_QUERY = 0,                    // 16*256
  WS_U     = WS_QUERY + BB*HIDDEN, // 16*256
  WS_V     = WS_U     + BB*HIDDEN, // 4*256
  WS_T     = WS_V     + 4*HIDDEN,  // 256
  WS_P     = WS_T     + HIDDEN,    // 256
  WS_S     = WS_P     + HIDDEN,    // 256
  WS_CN    = WS_S     + HIDDEN,    // 16
  WS_FLOATS = WS_CN + BB
};
#define WS_MBF_BYTES (((WS_FLOATS * 4) + 255) & ~255)   // bf16 M starts here

// ============================================================
// Kernel 1: query = center @ Wq^T + bq ; cnorm[b] = max(||center_b||,1e-8)
// grid 16, block 256
// ============================================================
__global__ __launch_bounds__(256) void k_query(const float* __restrict__ center,
                                               const float* __restrict__ Wq,
                                               const float* __restrict__ bq,
                                               float* __restrict__ ws) {
  __shared__ float ssq[256];
  int b = blockIdx.x, t = threadIdx.x;
  const float* c = center + b * TEXT_DIM;
  const float* w = Wq + t * TEXT_DIM;
  float acc = 0.f;
  for (int d = 0; d < TEXT_DIM; ++d) acc += w[d] * c[d];
  ws[WS_QUERY + b * HIDDEN + t] = acc + bq[t];
  float sp = 0.f;
  for (int i = 0; i < 3; ++i) { float cv = c[t * 3 + i]; sp += cv * cv; }
  ssq[t] = sp; __syncthreads();
  for (int s = 128; s > 0; s >>= 1) { if (t < s) ssq[t] += ssq[t + s]; __syncthreads(); }
  if (t == 0) ws[WS_CN + b] = fmaxf(sqrtf(ssq[0]), 1e-8f);
}

// ============================================================
// Kernel 2: per-batch bias u_b, edge vectors, time/prior/sem vectors
// grid 17, block 256. blocks 0..15 -> u_b ; block 16 -> v/t/p/s
// W1 is (256 x 1281): cols [0:256)=query [256:512)=cand [512:768)=edge
//                          [768:1024)=time [1024:1280)=prior [1280]=sem
// ============================================================
__global__ __launch_bounds__(256) void k_vecs(const float* __restrict__ W1,
                                              const float* __restrict__ b1,
                                              const float* __restrict__ bc,
                                              const float* __restrict__ bt,
                                              const float* __restrict__ bs,
                                              const float* __restrict__ Ee,
                                              const float* __restrict__ Wt,
                                              const float* __restrict__ Wsp,
                                              float* __restrict__ ws) {
  int t = threadIdx.x;
  const float* row = W1 + (size_t)t * 1281;
  if (blockIdx.x < BB) {
    int b = blockIdx.x;
    const float* q = ws + WS_QUERY + b * HIDDEN;
    float acc = b1[t];
    for (int k = 0; k < 256; ++k) acc += row[k]        * q[k];
    for (int k = 0; k < 256; ++k) acc += row[256 + k]  * bc[k];
    for (int k = 0; k < 256; ++k) acc += row[768 + k]  * bt[k];
    for (int k = 0; k < 256; ++k) acc += row[1024 + k] * bs[k];
    ws[WS_U + b * HIDDEN + t] = acc;
  } else {
    for (int e = 0; e < 4; ++e) {
      float a = 0.f;
      for (int k = 0; k < 256; ++k) a += row[512 + k] * Ee[e * 256 + k];
      ws[WS_V + e * 256 + t] = a;
    }
    float at = 0.f, ap = 0.f;
    for (int k = 0; k < 256; ++k) { at += row[768 + k] * Wt[k]; ap += row[1024 + k] * Wsp[k]; }
    ws[WS_T + t] = at; ws[WS_P + t] = ap; ws[WS_S + t] = row[1280];
  }
}

// ============================================================
// Kernel 3: M[h][d] = sum_k W1[h][256+k] * Wc[k][d]  -> bf16
// grid 256 (h), block 256 (d/3)
// ============================================================
__global__ __launch_bounds__(256) void k_mmat(const float* __restrict__ W1,
                                              const float* __restrict__ Wc,
                                              unsigned short* __restrict__ Mbf) {
  int h = blockIdx.x, t = threadIdx.x;
  const float* row = W1 + (size_t)h * 1281 + 256;
  float a0 = 0.f, a1 = 0.f, a2 = 0.f;
  for (int k = 0; k < 256; ++k) {
    float w = row[k];
    const float* wc = Wc + (size_t)k * TEXT_DIM;
    a0 += w * wc[t]; a1 += w * wc[t + 256]; a2 += w * wc[t + 512];
  }
  unsigned short* out = Mbf + (size_t)h * TEXT_DIM;
  out[t] = f2bf(a0); out[t + 256] = f2bf(a1); out[t + 512] = f2bf(a2);
}

// ============================================================
// Kernel 4: main WMMA kernel.
// 64 candidates per WG (dynamic LDS ~110 KB), 256 threads = 8 waves.
// Wave w owns cols [32w,32w+32) (2 N-tiles) x 4 M-tiles -> 8 accum tiles.
// Each B fragment is reused by 4 WMMAs => half the L2 traffic of the
// 32-row variant; keeps the kernel under the 23.3 TB/s HBM roofline.
// K-loop: 24 x K=32 -> 192 v_wmma_f32_16x16x32_bf16 per wave.
// ============================================================
#define ROWS 64
#define ASTR 776   // padded bf16 row stride (+8) -> bank rotation for ds_load_b128
#define ATILE_BYTES (ROWS * ASTR * 2)                     // 99328
#define SM_FLOATS   (5*256 + 4*256 + 6*ROWS)              // u,t,p,s,w2 + v + row scalars
#define SMEM_BYTES  (ATILE_BYTES + SM_FLOATS*4 + 2*ROWS*4)

__global__ __launch_bounds__(256) void k_main(
    const float* __restrict__ cand, const float* __restrict__ center,
    const int* __restrict__ maskp,  const int* __restrict__ etype,
    const float* __restrict__ year, const float* __restrict__ cache,
    const float* __restrict__ W2,   const float* __restrict__ b2,
    const float* __restrict__ ws,   const unsigned short* __restrict__ Mbf,
    float* __restrict__ out) {
  extern __shared__ __align__(16) char smem[];
  unsigned short* Atile = (unsigned short*)smem;
  float* uS     = (float*)(smem + ATILE_BYTES);
  float* tS     = uS  + 256;
  float* pS     = tS  + 256;
  float* sS     = pS  + 256;
  float* w2S    = sS  + 256;
  float* vS     = w2S + 256;          // 4*256
  float* yearS  = vS + 4*256;
  float* cacheS = yearS  + ROWS;
  float* semS   = cacheS + ROWS;
  float* dotS   = semS   + ROWS;
  float* sqS    = dotS   + ROWS;
  float* logitS = sqS    + ROWS;
  int*   edgeS  = (int*)(logitS + ROWS);
  int*   maskS  = edgeS + ROWS;

  int t = threadIdx.x;
  int base = blockIdx.x * ROWS;       // global candidate row base
  int b = base >> 12;                 // batch index (ROWS | 4096)

  // stage per-batch / shared vectors
  uS[t]  = ws[WS_U + b * 256 + t];
  tS[t]  = ws[WS_T + t];  pS[t] = ws[WS_P + t];  sS[t] = ws[WS_S + t];
  w2S[t] = W2[t];
  for (int e = 0; e < 4; ++e) vS[e * 256 + t] = ws[WS_V + e * 256 + t];
  if (t < ROWS) {
    logitS[t] = 0.f;
    yearS[t]  = year[base + t];
    cacheS[t] = cache[base + t];
    int e = etype[base + t]; e = e < 0 ? 0 : (e > 3 ? 3 : e);
    edgeS[t]  = e;
    maskS[t]  = maskp[base + t];
  }
  float cnorm = ws[WS_CN + b];

  // stream 64x768 fp32 -> bf16 LDS tile; fuse dot(center,x) and ||x||^2 (fp32 exact)
  {
    int r = t >> 2, c = t & 3;        // 4 threads per row, 192 floats each
    const float4* xs = (const float4*)(cand + (size_t)(base + r) * TEXT_DIM + c * 192);
    const float4* cs = (const float4*)(center + (size_t)b * TEXT_DIM + c * 192);
    unsigned short* dst = &Atile[r * ASTR + c * 192];
    float dp = 0.f, sp = 0.f;
    #pragma unroll 4
    for (int i = 0; i < 48; ++i) {
      float4 x = xs[i]; float4 cc = cs[i];
      dp += x.x * cc.x + x.y * cc.y + x.z * cc.z + x.w * cc.w;
      sp += x.x * x.x + x.y * x.y + x.z * x.z + x.w * x.w;
      dst[i * 4 + 0] = f2bf(x.x); dst[i * 4 + 1] = f2bf(x.y);
      dst[i * 4 + 2] = f2bf(x.z); dst[i * 4 + 3] = f2bf(x.w);
    }
    for (int off = 1; off < 4; off <<= 1) {   // reduce over the 4 lanes of this row group
      dp += __shfl_xor(dp, off, 32);
      sp += __shfl_xor(sp, off, 32);
    }
    if (c == 0) { dotS[r] = dp; sqS[r] = sp; }
  }
  __syncthreads();
  if (t < ROWS) semS[t] = dotS[t] / (cnorm * fmaxf(sqrtf(sqS[t]), 1e-8f));
  __syncthreads();

  int wave = t >> 5, lane = t & 31;
  int h0 = wave * 32;                 // this wave's 32 output columns
  int lmod = lane & 15, lhalf = lane >> 4;

  v8f zero = {0.f, 0.f, 0.f, 0.f, 0.f, 0.f, 0.f, 0.f};
  v8f acc[4][2];
  #pragma unroll
  for (int mt = 0; mt < 4; ++mt) { acc[mt][0] = zero; acc[mt][1] = zero; }

  for (int ko = 0; ko < TEXT_DIM; ko += 32) {
    BF16x16 afr[4], bfr[2];
    // A (16x32 bf16): lane<16 -> K {ko..ko+7, ko+16..ko+23}; lane>=16 -> +8
    #pragma unroll
    for (int mt = 0; mt < 4; ++mt) {
      const unsigned short* rp = &Atile[(mt * 16 + lmod) * ASTR + ko + lhalf * 8];
      afr[mt].q[0] = *(const u32x4*)(rp);
      afr[mt].q[1] = *(const u32x4*)(rp + 16);
    }
    // B (32x16 bf16): lane -> column h0+nt*16+lmod; K {ko..ko+7, ko+16..ko+23} (+8 for lane>=16) from Mbf row
    #pragma unroll
    for (int nt = 0; nt < 2; ++nt) {
      const unsigned short* mp = Mbf + (size_t)(h0 + nt * 16 + lmod) * TEXT_DIM + ko + lhalf * 8;
      bfr[nt].q[0] = *(const u32x4*)(mp);
      bfr[nt].q[1] = *(const u32x4*)(mp + 16);
    }
    #pragma unroll
    for (int mt = 0; mt < 4; ++mt)
      #pragma unroll
      for (int nt = 0; nt < 2; ++nt)
        acc[mt][nt] = __builtin_amdgcn_wmma_f32_16x16x32_bf16(
            false, afr[mt].v, false, bfr[nt].v, (short)0, acc[mt][nt], false, false);
  }

  // epilogue: rank-1 terms + exact GELU + dot with W2, reduce per row
  #pragma unroll
  for (int mt = 0; mt < 4; ++mt) {
    #pragma unroll
    for (int nt = 0; nt < 2; ++nt) {
      #pragma unroll
      for (int j = 0; j < 8; ++j) {
        int r = mt * 16 + j + lhalf * 8;          // tile C layout: M = j (+8 for upper half)
        int h = h0 + nt * 16 + lmod;              // N = lane&15
        float val = acc[mt][nt][j] + uS[h] + vS[edgeS[r] * 256 + h]
                  + yearS[r] * tS[h] + cacheS[r] * pS[h] + semS[r] * sS[h];
        float g = 0.5f * val * (1.f + erff(val * 0.70710678f));
        float contrib = g * w2S[h];
        for (int off = 1; off < 16; off <<= 1)    // butterfly within each 16-lane half
          contrib += __shfl_xor(contrib, off, 32);
        if (lmod == 0) atomicAdd(&logitS[r], contrib);
      }
    }
  }
  __syncthreads();
  if (t < ROWS) {
    float lg = logitS[t] + b2[0];
    if (!maskS[t]) lg = -1e9f;
    out[2 * BB * TOPK + base + t] = lg;           // logits section at offset 2048
  }
}

// ============================================================
// Kernel 5: per-batch top-64 (iterative argmax in LDS). grid 16, block 256.
// d_out layout: [0,1024) idx (as float), [1024,2048) top logits, [2048,...) logits
// ============================================================
__global__ __launch_bounds__(256) void k_topk(float* __restrict__ out) {
  __shared__ float vals[NN];
  __shared__ float rv[256];
  __shared__ int   ri[256];
  int b = blockIdx.x, t = threadIdx.x;
  float* lg = out + 2 * BB * TOPK + (size_t)b * NN;
  for (int i = t; i < NN; i += 256) vals[i] = lg[i];
  __syncthreads();
  for (int k = 0; k < TOPK; ++k) {
    float bv = -INFINITY; int bi = 0x7FFFFFFF;
    for (int i = t; i < NN; i += 256) {
      float v = vals[i];
      if (v > bv) { bv = v; bi = i; }             // strict > keeps smallest index on ties
    }
    rv[t] = bv; ri[t] = bi;
    __syncthreads();
    for (int s = 128; s > 0; s >>= 1) {
      if (t < s) {
        if (rv[t + s] > rv[t] || (rv[t + s] == rv[t] && ri[t + s] < ri[t])) {
          rv[t] = rv[t + s]; ri[t] = ri[t + s];
        }
      }
      __syncthreads();
    }
    if (t == 0) {
      out[b * TOPK + k]             = (float)ri[0];
      out[BB * TOPK + b * TOPK + k] = rv[0];
      vals[ri[0]] = -INFINITY;
    }
    __syncthreads();
  }
}

// ============================================================
extern "C" void kernel_launch(void* const* d_in, const int* in_sizes, int n_in,
                              void* d_out, int out_size, void* d_ws, size_t ws_size,
                              hipStream_t stream) {
  (void)in_sizes; (void)n_in; (void)out_size; (void)ws_size;
  const float* center = (const float*)d_in[0];
  const float* cand   = (const float*)d_in[1];
  const int*   maskp  = (const int*)d_in[2];
  const int*   etype  = (const int*)d_in[3];
  const float* year   = (const float*)d_in[4];
  const float* cache  = (const float*)d_in[5];
  // d_in[6] = top_k (fixed 64 per setup)
  const float* Wq  = (const float*)d_in[7];
  const float* bq  = (const float*)d_in[8];
  const float* Wc  = (const float*)d_in[9];
  const float* bc  = (const float*)d_in[10];
  const float* Ee  = (const float*)d_in[11];
  const float* Wt  = (const float*)d_in[12];
  const float* bt  = (const float*)d_in[13];
  const float* Wsp = (const float*)d_in[14];
  const float* bs  = (const float*)d_in[15];
  const float* W1  = (const float*)d_in[16];
  const float* b1  = (const float*)d_in[17];
  const float* W2  = (const float*)d_in[18];
  const float* b2  = (const float*)d_in[19];

  float* ws = (float*)d_ws;
  unsigned short* Mbf = (unsigned short*)((char*)d_ws + WS_MBF_BYTES);
  float* out = (float*)d_out;

  k_query<<<BB, 256, 0, stream>>>(center, Wq, bq, ws);
  k_vecs <<<BB + 1, 256, 0, stream>>>(W1, b1, bc, bt, bs, Ee, Wt, Wsp, ws);
  k_mmat <<<HIDDEN, 256, 0, stream>>>(W1, Wc, Mbf);
  k_main <<<(BB * NN) / ROWS, 256, SMEM_BYTES, stream>>>(cand, center, maskp, etype, year,
                                                         cache, W2, b2, ws, Mbf, out);
  k_topk <<<BB, 256, 0, stream>>>(out);
}
